// SpatialRewardModel_29557964931640
// MI455X (gfx1250) — matrix operand order repaired and checked by reading, not compiled, not run on hardware
//
#include <hip/hip_runtime.h>
#include <hip/hip_bf16.h>

typedef __attribute__((ext_vector_type(16))) _Float16 v16h;
typedef __attribute__((ext_vector_type(8)))  _Float16 v8h;
typedef __attribute__((ext_vector_type(8)))  float    v8f;

#define B_SZ 32768
#define NBOX 64
#define SPAT 256
#define TXT  768
#define FEAT 1024
#define HID  768
#define HID2 384

// ---------------------------------------------------------------------------
// Weight convert + transpose: W[k][n] (f32, row-major KxN) -> Wt[n][k] (f16)
// ---------------------------------------------------------------------------
__global__ void convert_wt(const float* __restrict__ W, _Float16* __restrict__ Wt,
                           int K, int N) {
    int idx = blockIdx.x * blockDim.x + threadIdx.x;
    if (idx < K * N) {
        int k = idx / N, n = idx % N;
        Wt[(size_t)n * K + k] = (_Float16)W[idx];
    }
}

// ---------------------------------------------------------------------------
// Build X[b][0:256] = spatial (closed-form last-write-wins winners),
//       X[b][256:1024] = text, all converted to f16.
// One block (256 threads) per batch element.
// ---------------------------------------------------------------------------
__global__ __launch_bounds__(256) void build_x(const float* __restrict__ boxes,
                                               const float* __restrict__ scores,
                                               const float* __restrict__ text,
                                               _Float16* __restrict__ X) {
    __shared__ float cx[NBOX], cy[NBOX], sc[NBOX];
    int b = blockIdx.x;
    int t = threadIdx.x;
    if (t < NBOX) {
        const float* bx = boxes + (size_t)b * NBOX * 4 + t * 4;
        cx[t] = (bx[0] + bx[2]) * 0.5f;
        cy[t] = (bx[1] + bx[3]) * 0.5f;
        sc[t] = scores[(size_t)b * NBOX + t];
    }
    __syncthreads();

    // slot t: j = t&63, q = t>>6; winner i = largest i<j with i%4==q
    int j = t & 63;
    int q = t >> 6;
    float val = 0.0f;
    if (j >= 1 && q <= j - 1) {
        int i = (j - 1) - ((j - 1 - q) & 3);
        float dx = cx[i] - cx[j];
        float dy = cy[i] - cy[j];
        val = sqrtf(dx * dx + dy * dy) * sc[i] * sc[j];
    }
    _Float16* Xr = X + (size_t)b * FEAT;
    Xr[t] = (_Float16)val;
    const float* tr = text + (size_t)b * TXT;
    for (int u = t; u < TXT; u += 256) Xr[SPAT + u] = (_Float16)tr[u];
}

// ---------------------------------------------------------------------------
// WMMA GEMM + bias + ReLU:  C[M x N] = relu(A[M x K] @ Bt[N x K]^T + bias)
// Per wave: 4 M-frags x NF N-frags (12 v8f accumulators). Block covers full N.
// A layout (16-bit, 16x32): lane<16 -> K {k0+hi*8..+7, k0+16+hi*8..+7}
// B layout (16-bit, 32x16): lane group hi -> K {k0+hi*8..+7, k0+16+hi*8..+7}
// ---------------------------------------------------------------------------
template <int K, int N, int NWAVES>
__global__ __launch_bounds__(NWAVES * 32) void gemm_relu_f16(
    const _Float16* __restrict__ A,    // [M x K] row-major f16
    const _Float16* __restrict__ Bt,   // [N x K] row-major f16 (pre-transposed)
    const float*    __restrict__ bias, // [N] f32
    _Float16*       __restrict__ C)    // [M x N] row-major f16
{
    constexpr int MF = 4;
    constexpr int NF = N / (NWAVES * 16);
    static_assert(N == NWAVES * NF * 16, "tile mismatch");

    const int wave  = threadIdx.x >> 5;
    const int lane  = threadIdx.x & 31;
    const int l15   = lane & 15;
    const int hi    = lane >> 4;
    const int mbase = blockIdx.x * (MF * 16);
    const int nwave = wave * (NF * 16);

    v8f acc[MF][NF];
#pragma unroll
    for (int mf = 0; mf < MF; ++mf)
#pragma unroll
        for (int nf = 0; nf < NF; ++nf) acc[mf][nf] = (v8f)0.0f;

    const _Float16* Arow[MF];
#pragma unroll
    for (int mf = 0; mf < MF; ++mf)
        Arow[mf] = A + (size_t)(mbase + mf * 16 + l15) * K;
    const _Float16* Brow[NF];
#pragma unroll
    for (int nf = 0; nf < NF; ++nf)
        Brow[nf] = Bt + (size_t)(nwave + nf * 16 + l15) * K;

    union Frag { v16h v; v8h h[2]; };

    for (int k0 = 0; k0 < K; k0 += 32) {
        Frag a[MF];
#pragma unroll
        for (int mf = 0; mf < MF; ++mf) {
            a[mf].h[0] = *(const v8h*)(Arow[mf] + k0 + hi * 8);
            a[mf].h[1] = *(const v8h*)(Arow[mf] + k0 + 16 + hi * 8);
        }
#pragma unroll
        for (int nf = 0; nf < NF; ++nf) {
            Frag bfr;
            bfr.h[0] = *(const v8h*)(Brow[nf] + k0 + hi * 8);
            bfr.h[1] = *(const v8h*)(Brow[nf] + k0 + 16 + hi * 8);
#pragma unroll
            for (int mf = 0; mf < MF; ++mf) {
                acc[mf][nf] = __builtin_amdgcn_wmma_f32_16x16x32_f16(
                    /*neg_a=*/false, a[mf].v, /*neg_b=*/false, bfr.v,
                    /*c_mod=*/(short)0, acc[mf][nf],
                    /*reuse_a=*/false, /*reuse_b=*/false);
            }
        }
    }

    // Epilogue: bias + ReLU + f16 store.  C layout: vgpr r -> M = r + hi*8.
#pragma unroll
    for (int nf = 0; nf < NF; ++nf) {
        const int n  = nwave + nf * 16 + l15;
        const float bv = bias[n];
#pragma unroll
        for (int mf = 0; mf < MF; ++mf) {
            const int m0 = mbase + mf * 16 + hi * 8;
#pragma unroll
            for (int r = 0; r < 8; ++r) {
                float v = acc[mf][nf][r] + bv;
                v = fmaxf(v, 0.0f);
                C[(size_t)(m0 + r) * N + n] = (_Float16)v;
            }
        }
    }
}

// ---------------------------------------------------------------------------
// Final layer: out[m] = dot(H2[m][0:384], W3) + b3. One wave per row.
// ---------------------------------------------------------------------------
__global__ __launch_bounds__(256) void final_dot(const _Float16* __restrict__ H2,
                                                 const float* __restrict__ W3,
                                                 const float* __restrict__ b3,
                                                 float* __restrict__ out, int M) {
    int row  = blockIdx.x * 8 + (threadIdx.x >> 5);
    int lane = threadIdx.x & 31;
    if (row >= M) return;
    const _Float16* h = H2 + (size_t)row * HID2;
    float s = 0.0f;
#pragma unroll
    for (int k = lane; k < HID2; k += 32) s += (float)h[k] * W3[k];
#pragma unroll
    for (int off = 16; off; off >>= 1) s += __shfl_xor(s, off, 32);
    if (lane == 0) out[row] = s + b3[0];
}

// ---------------------------------------------------------------------------
extern "C" void kernel_launch(void* const* d_in, const int* in_sizes, int n_in,
                              void* d_out, int out_size, void* d_ws, size_t ws_size,
                              hipStream_t stream) {
    (void)in_sizes; (void)n_in; (void)out_size; (void)ws_size;

    const float* boxes  = (const float*)d_in[0];  // [B,64,4]
    const float* scores = (const float*)d_in[1];  // [B,64]
    const float* text   = (const float*)d_in[2];  // [B,768]
    const float* W1     = (const float*)d_in[3];  // [1024,768]
    const float* b1     = (const float*)d_in[4];  // [768]
    const float* W2     = (const float*)d_in[5];  // [768,384]
    const float* b2     = (const float*)d_in[6];  // [384]
    const float* W3     = (const float*)d_in[7];  // [384]
    const float* b3     = (const float*)d_in[8];  // [1]
    float* out = (float*)d_out;

    // Workspace layout (H2 aliases X: X is dead once GEMM1 completes)
    char* ws = (char*)d_ws;
    size_t off = 0;
    _Float16* X   = (_Float16*)(ws + off);  off += (size_t)B_SZ * FEAT * 2;  // 64 MB
    _Float16* H1  = (_Float16*)(ws + off);  off += (size_t)B_SZ * HID  * 2;  // 48 MB
    _Float16* W1t = (_Float16*)(ws + off);  off += (size_t)HID  * FEAT * 2;  // 1.5 MB
    _Float16* W2t = (_Float16*)(ws + off);  off += (size_t)HID2 * HID  * 2;  // 0.6 MB
    _Float16* H2  = X;                                                       // alias

    // 1) weights -> f16 transposed
    convert_wt<<<(FEAT * HID + 255) / 256, 256, 0, stream>>>(W1, W1t, FEAT, HID);
    convert_wt<<<(HID * HID2 + 255) / 256, 256, 0, stream>>>(W2, W2t, HID, HID2);

    // 2) build combined f16 feature matrix X [B x 1024]
    build_x<<<B_SZ, 256, 0, stream>>>(boxes, scores, text, X);

    // 3) H1 = relu(X @ W1 + b1)   [B x 768],  16 waves/block cover N=768
    gemm_relu_f16<FEAT, HID, 16><<<B_SZ / 64, 512, 0, stream>>>(X, W1t, b1, H1);

    // 4) H2 = relu(H1 @ W2 + b2)  [B x 384],  8 waves/block cover N=384
    gemm_relu_f16<HID, HID2, 8><<<B_SZ / 64, 256, 0, stream>>>(H1, W2t, b2, H2);

    // 5) out = H2 @ W3 + b3
    final_dot<<<(B_SZ + 7) / 8, 256, 0, stream>>>(H2, W3, b3, out, B_SZ);
}
